// DotProductAttention_43353399886256
// MI455X (gfx1250) — matrix operand order repaired and checked by reading, not compiled, not run on hardware
//
#include <hip/hip_runtime.h>

// ---------------------------------------------------------------------------
// Flash-attention forward for B=8, Lq=Lk=2048, D=128, fp32 I/O, per-batch
// key-length mask.  bf16 WMMA (16x16x32) with f32 accumulation, f32 softmax.
// Wave32: each wave owns a 16-row query tile; 4 waves/block share K/V staging.
// Key block = 64.  Softmax denominator computed on the matrix engine as P*1.
// ---------------------------------------------------------------------------

typedef __attribute__((ext_vector_type(16))) __bf16 v16bf;
typedef __attribute__((ext_vector_type(8)))  __bf16 v8bf;
typedef __attribute__((ext_vector_type(4)))  __bf16 v4bf;
typedef __attribute__((ext_vector_type(8)))  float  v8f;
typedef __attribute__((ext_vector_type(4)))  float  v4f;

#define DH     128
#define BK     64
#define SCALE  0.08838834764831845f   // 1/sqrt(128)
#define MASKV  (-1.0e6f)

static __device__ __forceinline__ __bf16 f2bf(float f) {
    return (__bf16)f;                 // native v_cvt on gfx1250 (RNE)
}

static __device__ __forceinline__ v16bf join8(v8bf lo, v8bf hi) {
    return __builtin_shufflevector(lo, hi, 0,1,2,3,4,5,6,7,8,9,10,11,12,13,14,15);
}

__global__ __launch_bounds__(128)
void fa_fwd_bf16wmma(const float* __restrict__ Q, const float* __restrict__ Kg,
                     const float* __restrict__ Vg, const int* __restrict__ VL,
                     float* __restrict__ O, int B, int Lq, int Lk, int qpb)
{
    __shared__ __attribute__((aligned(16))) __bf16 Ks[BK * DH];      // K block, row-major [key][d]
    __shared__ __attribute__((aligned(16))) __bf16 Vt[DH * BK];      // V block transposed [d][key]
    __shared__ __attribute__((aligned(16))) __bf16 Pls[4][16 * BK];  // per-wave P staging [q][k]

    const int b    = blockIdx.x / qpb;
    const int qb   = blockIdx.x % qpb;
    const int w    = threadIdx.x >> 5;          // wave in block (0..3)
    const int ln   = threadIdx.x & 31;          // lane
    const int q0   = (qb * 4 + w) * 16;         // this wave's query tile base
    const int col  = ln & 15;                   // N / row index within 16-group
    const int hi16 = ln >> 4;                   // lane half
    const int kbA  = hi16 ? 8  : 0;             // A-matrix K-base for this lane half
    const int kbB  = hi16 ? 8  : 0;             // B-matrix K-base for this lane half

    const int valid = VL[b];                                  // 1..Lk
    const int kend  = ((valid + (BK - 1)) / BK) * BK;         // multiple of 64 (<= Lk)

    // ---- preload Q as 4 bf16 A-fragments (16 rows x 128 d) -----------------
    const int  qm   = (q0 + col < Lq) ? (q0 + col) : (Lq - 1);
    const float* qrow = Q + ((size_t)b * Lq + qm) * DH;
    v16bf qf[4];
#pragma unroll
    for (int c = 0; c < 4; ++c) {
        const float* p = qrow + c * 32 + kbA;
        v4f a0 = *(const v4f*)(p + 0);
        v4f a1 = *(const v4f*)(p + 4);
        v4f a2 = *(const v4f*)(p + 16);
        v4f a3 = *(const v4f*)(p + 20);
        v8f lo = __builtin_shufflevector(a0, a1, 0,1,2,3,4,5,6,7);
        v8f hi = __builtin_shufflevector(a2, a3, 0,1,2,3,4,5,6,7);
        qf[c] = join8(__builtin_convertvector(lo, v8bf),
                      __builtin_convertvector(hi, v8bf));
    }

    // all-ones B fragment for the P*1 row-sum matmul
    v16bf onesb;
#pragma unroll
    for (int i = 0; i < 16; ++i) onesb[i] = f2bf(1.0f);

    const v8f z8 = {};
    v8f acc[8];                      // O accumulator: 8 tiles of 16 d-columns
#pragma unroll
    for (int t = 0; t < 8; ++t) acc[t] = z8;
    v8f accl = z8;                   // row-sum accumulator (softmax denominator)
    float mrun[8];
#pragma unroll
    for (int r = 0; r < 8; ++r) mrun[r] = -__builtin_inff();

    // staging maps
    const int krow = threadIdx.x >> 1;          // K: key row 0..63
    const int kseg = (threadIdx.x & 1) * 64;    // K: 64-wide d segment
    const int vkey = (threadIdx.x & 15) * 4;    // V: 4 adjacent keys
    const int vd0  = (threadIdx.x >> 4) * 16;   // V: 16-wide d segment

    for (int k0 = 0; k0 < kend; k0 += BK) {
        __syncthreads();   // protect Ks/Vt reuse across iterations

        // ---- cooperative stage: K -> Ks (bf16 row-major) ------------------
        const float* kr = Kg + ((size_t)b * Lk + k0 + krow) * DH + kseg;
        const float* vr = Vg + ((size_t)b * Lk + k0 + vkey) * DH + vd0;
        if (k0 + BK < kend) {  // prefetch next block (global_prefetch_b8)
            __builtin_prefetch(kr + BK * DH, 0, 1);
            __builtin_prefetch(vr + BK * DH, 0, 1);
        }
#pragma unroll
        for (int i = 0; i < 64; i += 8) {
            v4f ka = *(const v4f*)(kr + i);
            v4f kb = *(const v4f*)(kr + i + 4);
            v8f kk = __builtin_shufflevector(ka, kb, 0,1,2,3,4,5,6,7);
            *(v8bf*)(&Ks[krow * DH + kseg + i]) = __builtin_convertvector(kk, v8bf);
        }
        // ---- cooperative stage: V -> Vt (bf16, transposed [d][key]) -------
        // each thread owns 4 adjacent keys x 16 d; packed ds_store_b64
#pragma unroll
        for (int c2 = 0; c2 < 4; ++c2) {
            v4f r0 = *(const v4f*)(vr + 0 * DH + c2 * 4);
            v4f r1 = *(const v4f*)(vr + 1 * DH + c2 * 4);
            v4f r2 = *(const v4f*)(vr + 2 * DH + c2 * 4);
            v4f r3 = *(const v4f*)(vr + 3 * DH + c2 * 4);
#pragma unroll
            for (int e = 0; e < 4; ++e) {
                v4f pk = { r0[e], r1[e], r2[e], r3[e] };
                *(v4bf*)(&Vt[(vd0 + c2 * 4 + e) * BK + vkey]) =
                    __builtin_convertvector(pk, v4bf);
            }
        }
        __syncthreads();

        // ---- S = (Q K^T) for four 16-key tiles: 16 x v_wmma_f32_16x16x32_bf16
        v8f s0 = z8, s1 = z8, s2 = z8, s3 = z8;
#pragma unroll
        for (int c = 0; c < 4; ++c) {
            const __bf16* bp0 = &Ks[(col)      * DH + c * 32 + kbB];
            const __bf16* bp1 = &Ks[(16 + col) * DH + c * 32 + kbB];
            const __bf16* bp2 = &Ks[(32 + col) * DH + c * 32 + kbB];
            const __bf16* bp3 = &Ks[(48 + col) * DH + c * 32 + kbB];
            v16bf b0 = join8(*(const v8bf*)bp0, *(const v8bf*)(bp0 + 16));
            v16bf b1 = join8(*(const v8bf*)bp1, *(const v8bf*)(bp1 + 16));
            v16bf b2 = join8(*(const v8bf*)bp2, *(const v8bf*)(bp2 + 16));
            v16bf b3 = join8(*(const v8bf*)bp3, *(const v8bf*)(bp3 + 16));
            s0 = __builtin_amdgcn_wmma_f32_16x16x32_bf16(false, qf[c], false, b0,
                                                         (short)0, s0, false, false);
            s1 = __builtin_amdgcn_wmma_f32_16x16x32_bf16(false, qf[c], false, b1,
                                                         (short)0, s1, false, false);
            s2 = __builtin_amdgcn_wmma_f32_16x16x32_bf16(false, qf[c], false, b2,
                                                         (short)0, s2, false, false);
            s3 = __builtin_amdgcn_wmma_f32_16x16x32_bf16(false, qf[c], false, b3,
                                                         (short)0, s3, false, false);
        }

        // ---- scale + length mask + online softmax (max via lane shuffles) -
        const bool ok0 = (k0 + col)      < valid;
        const bool ok1 = (k0 + 16 + col) < valid;
        const bool ok2 = (k0 + 32 + col) < valid;
        const bool ok3 = (k0 + 48 + col) < valid;
        const int  prow = hi16 * 8;
        float alpha[8];
#pragma unroll
        for (int r = 0; r < 8; ++r) {
            float v0 = ok0 ? s0[r] * SCALE : MASKV;
            float v1 = ok1 ? s1[r] * SCALE : MASKV;
            float v2 = ok2 ? s2[r] * SCALE : MASKV;
            float v3 = ok3 ? s3[r] * SCALE : MASKV;
            // row max over the 64 key columns (16-lane half owns one row set)
            float rm = fmaxf(fmaxf(v0, v1), fmaxf(v2, v3));
            rm = fmaxf(rm, __shfl_xor(rm, 1));
            rm = fmaxf(rm, __shfl_xor(rm, 2));
            rm = fmaxf(rm, __shfl_xor(rm, 4));
            rm = fmaxf(rm, __shfl_xor(rm, 8));
            float mn = fmaxf(mrun[r], rm);
            alpha[r] = __expf(mrun[r] - mn);          // exp(-inf)=0 on first block
            mrun[r]  = mn;
            // stage P (C-layout -> row-major LDS) for the A-fragment reload
            __bf16* pr = &Pls[w][(prow + r) * BK + col];
            pr[0]  = f2bf(__expf(v0 - mn));
            pr[16] = f2bf(__expf(v1 - mn));
            pr[32] = f2bf(__expf(v2 - mn));
            pr[48] = f2bf(__expf(v3 - mn));
        }
#pragma unroll
        for (int r = 0; r < 8; ++r) {
            accl[r] *= alpha[r];
#pragma unroll
            for (int t = 0; t < 8; ++t) acc[t][r] *= alpha[r];
        }

        __syncthreads();   // also orders the per-wave Pls store->load

        // ---- O += P * V and rowsum += P * 1 : 18 WMMAs --------------------
        const __bf16* pbase = &Pls[w][col * BK];
        v16bf af0 = join8(*(const v8bf*)(pbase + kbA),      *(const v8bf*)(pbase + kbA + 16));
        v16bf af1 = join8(*(const v8bf*)(pbase + 32 + kbA), *(const v8bf*)(pbase + 32 + kbA + 16));
        accl = __builtin_amdgcn_wmma_f32_16x16x32_bf16(false, af0, false, onesb,
                                                       (short)0, accl, false, false);
        accl = __builtin_amdgcn_wmma_f32_16x16x32_bf16(false, af1, false, onesb,
                                                       (short)0, accl, false, false);
#pragma unroll
        for (int t = 0; t < 8; ++t) {
            const __bf16* vp = &Vt[(t * 16 + col) * BK];
            v16bf vb0 = join8(*(const v8bf*)(vp + kbB),      *(const v8bf*)(vp + kbB + 16));
            v16bf vb1 = join8(*(const v8bf*)(vp + 32 + kbB), *(const v8bf*)(vp + 32 + kbB + 16));
            acc[t] = __builtin_amdgcn_wmma_f32_16x16x32_bf16(false, af0, false, vb0,
                                                             (short)0, acc[t], false, false);
            acc[t] = __builtin_amdgcn_wmma_f32_16x16x32_bf16(false, af1, false, vb1,
                                                             (short)0, acc[t], false, false);
        }
    }

    // ---- epilogue: divide by row sums, store fp32 from C-layout ------------
    if (q0 < Lq) {
        const int row0 = hi16 * 8;
#pragma unroll
        for (int r = 0; r < 8; ++r) {
            float inv = 1.0f / accl[r];
            float* op = O + ((size_t)b * Lq + q0 + row0 + r) * DH + col;
#pragma unroll
            for (int t = 0; t < 8; ++t)
                op[t * 16] = acc[t][r] * inv;
        }
    }
}

extern "C" void kernel_launch(void* const* d_in, const int* in_sizes, int n_in,
                              void* d_out, int out_size, void* d_ws, size_t ws_size,
                              hipStream_t stream) {
    const float* Q  = (const float*)d_in[0];
    const float* K  = (const float*)d_in[1];
    const float* V  = (const float*)d_in[2];
    const int*   VL = (const int*)d_in[3];
    float* Out = (float*)d_out;

    const int B  = in_sizes[3];
    const int Lq = in_sizes[0] / (B * DH);
    const int Lk = in_sizes[1] / (B * DH);
    const int qpb = (Lq + 63) / 64;          // 64 query rows per block (4 waves x 16)

    dim3 grid(B * qpb), block(128);
    fa_fwd_bf16wmma<<<grid, block, 0, stream>>>(Q, K, V, VL, Out, B, Lq, Lk, qpb);
}
